// GCNO_22574348108066
// MI455X (gfx1250) — matrix operand order repaired and checked by reading, not compiled, not run on hardware
//
#include <hip/hip_runtime.h>
#include <hip/hip_bf16.h>

#define NN 100000
#define NE 3200000
#define IND 128
#define HID 256
#define NCLS 10
#define NGR 64

typedef __attribute__((ext_vector_type(16))) _Float16 v16h;
typedef __attribute__((ext_vector_type(8)))  float    v8f;

// ---------------- degree / norm ----------------
__global__ void k_deg_init(float* deg, int n) {
    int i = blockIdx.x * blockDim.x + threadIdx.x;
    if (i < n) deg[i] = 1.0f;                       // self loop
}
__global__ void k_deg_acc(const int* __restrict__ dst, float* deg, int e) {
    int i = blockIdx.x * blockDim.x + threadIdx.x;
    if (i < e) atomicAdd(&deg[dst[i]], 1.0f);
}
__global__ void k_dinv(const float* __restrict__ deg, float* dinv, int n) {
    int i = blockIdx.x * blockDim.x + threadIdx.x;
    if (i < n) dinv[i] = rsqrtf(deg[i]);
}
__global__ void k_norm(const int* __restrict__ src, const int* __restrict__ dst,
                       const float* __restrict__ dinv, float* norm, int e) {
    int i = blockIdx.x * blockDim.x + threadIdx.x;
    if (i < e) norm[i] = dinv[src[i]] * dinv[dst[i]];
}

// ---------------- f32 -> f16 (optional relu) ----------------
__global__ void k_tohalf(const float* __restrict__ in, _Float16* __restrict__ out,
                         size_t n, int do_relu) {
    size_t i = (size_t)blockIdx.x * blockDim.x + threadIdx.x;
    if (i < n) {
        float v = in[i];
        if (do_relu) v = fmaxf(v, 0.0f);
        out[i] = (_Float16)v;
    }
}

// ---------------- pack W (f32 [K,256]) into WMMA-B lane-contiguous f16 ----------------
// Bp[(kc*16 + tile)*32 + lane][0..15] = W[kc*32 + (lane>>4)*16 + j][tile*16 + (lane&15)]
__global__ void k_packB(const float* __restrict__ W, _Float16* __restrict__ Bp, int K) {
    int idx = blockIdx.x * blockDim.x + threadIdx.x;
    int total = (K >> 5) * 16 * 32;
    if (idx >= total) return;
    int lane = idx & 31;
    int tile = (idx >> 5) & 15;
    int kc   = idx >> 9;
    int n  = tile * 16 + (lane & 15);
    int k0 = kc * 32 + (lane >> 4) * 16;
    _Float16* dst = Bp + (size_t)idx * 16;
#pragma unroll
    for (int j = 0; j < 16; ++j)
        dst[j] = (_Float16)W[(size_t)(k0 + j) * 256 + n];
}

// ---------------- WMMA GEMM: H[N,256] = Xh[N,K] * W ----------------
// block = 4 waves; wave w owns col tiles [4w,4w+4); grid.x = N/16 row tiles.
__global__ __launch_bounds__(128)
void k_gemm_wmma(const _Float16* __restrict__ Xh, const _Float16* __restrict__ Bp,
                 float* __restrict__ H, int K) {
    const int wave = threadIdx.x >> 5;
    const int lane = threadIdx.x & 31;
    const int row0 = blockIdx.x * 16;
    const int m    = lane & 15;
    const int koff = (lane >> 4) * 16;    // A layout: lanes 16-31 hold K+16..K+31

    v8f c0 = {}, c1 = {}, c2 = {}, c3 = {};
    const int kchunks = K >> 5;
    const _Float16* arow = Xh + (size_t)(row0 + m) * K;

    for (int kc = 0; kc < kchunks; ++kc) {
        const int kbase = kc << 5;
        union { uint4 u[2]; v16h v; } au;
        au.u[0] = *(const uint4*)(arow + kbase + koff);        // K 0..7 (or 16..23)
        au.u[1] = *(const uint4*)(arow + kbase + 8 + koff);    // K 8..15 (or 24..31)
        const v16h a = au.v;

        const _Float16* bbase = Bp + (((size_t)(kc * 16 + wave * 4) * 32) + lane) * 16;
#pragma unroll
        for (int t = 0; t < 4; ++t) {
            union { uint4 u[2]; v16h v; } bu;
            const _Float16* bp = bbase + (size_t)t * 32 * 16;
            bu.u[0] = *(const uint4*)(bp);
            bu.u[1] = *(const uint4*)(bp + 8);
            v8f* cp = (t == 0) ? &c0 : (t == 1) ? &c1 : (t == 2) ? &c2 : &c3;
            *cp = __builtin_amdgcn_wmma_f32_16x16x32_f16(
                      false, a, false, bu.v, (short)0, *cp, false, false);
        }
    }

    const int ncol = lane & 15;
    const int mrow = row0 + (lane >> 4) * 8;
#pragma unroll
    for (int t = 0; t < 4; ++t) {
        const v8f c = (t == 0) ? c0 : (t == 1) ? c1 : (t == 2) ? c2 : c3;
        const int n0 = (wave * 4 + t) * 16 + ncol;
#pragma unroll
        for (int g = 0; g < 8; ++g)
            H[(size_t)(mrow + g) * 256 + n0] = c[g];
    }
}

// ---------------- agg init: self-loop + bias ----------------
__global__ __launch_bounds__(256)
void k_agg_init(const float* __restrict__ H, const float* __restrict__ dinv,
                const float* __restrict__ b, float* __restrict__ agg) {
    size_t idx = (size_t)blockIdx.x * 256 + threadIdx.x;
    int i = (int)(idx >> 8);
    int f = (int)(idx & 255);
    float di = dinv[i];
    agg[idx] = di * di * H[idx] + b[f];
}

// ---------------- edge scatter: agg[dst] += norm*H[src] ----------------
__global__ __launch_bounds__(64)
void k_edge_agg(const int* __restrict__ src, const int* __restrict__ dst,
                const float* __restrict__ norm, const float* __restrict__ H,
                float* __restrict__ agg) {
    int e = blockIdx.x;
    int s = src[e], d = dst[e];
    float w = norm[e];
    int f = threadIdx.x * 4;
    const float4 hv = *(const float4*)(H + (size_t)s * 256 + f);
    float* ap = agg + (size_t)d * 256 + f;
    atomicAdd(ap + 0, w * hv.x);
    atomicAdd(ap + 1, w * hv.y);
    atomicAdd(ap + 2, w * hv.z);
    atomicAdd(ap + 3, w * hv.w);
}

// ---------------- pooling ----------------
__global__ void k_pool_zero(float* sums, float* cnt) {
    int i = blockIdx.x * blockDim.x + threadIdx.x;
    if (i < NGR * HID) sums[i] = 0.0f;
    if (i < NGR) cnt[i] = 0.0f;
}
// block handles 128 nodes; thread t owns feature t; batch is sorted -> few flushes
__global__ __launch_bounds__(256)
void k_pool_acc(const float* __restrict__ agg, const int* __restrict__ batch,
                float* __restrict__ sums, float* __restrict__ cnt, int n) {
    int f  = threadIdx.x;
    int i0 = blockIdx.x * 128;
    int i1 = i0 + 128; if (i1 > n) i1 = n;
    if (i0 >= n) return;
    int cur = batch[i0];
    float acc = 0.0f, c = 0.0f;
    for (int i = i0; i < i1; ++i) {
        int g = batch[i];
        if (g != cur) {
            atomicAdd(&sums[cur * HID + f], acc);
            if (f == 0) atomicAdd(&cnt[cur], c);
            acc = 0.0f; c = 0.0f; cur = g;
        }
        acc += agg[(size_t)i * HID + f];
        c += 1.0f;
    }
    atomicAdd(&sums[cur * HID + f], acc);
    if (f == 0) atomicAdd(&cnt[cur], c);
}

__global__ void k_final(const float* __restrict__ sums, const float* __restrict__ cnt,
                        const float* __restrict__ Wlin, const float* __restrict__ blin,
                        float* __restrict__ out) {
    int idx = blockIdx.x * blockDim.x + threadIdx.x;
    if (idx >= NGR * NCLS) return;
    int g = idx / NCLS, cl = idx % NCLS;
    float cc = cnt[g];
    float acc = 0.0f;
    if (cc > 0.0f) {
        float inv = 1.0f / fmaxf(cc, 1.0f);
        for (int f = 0; f < HID; ++f)
            acc += (sums[g * HID + f] * inv) * Wlin[f * NCLS + cl];
    }
    out[idx] = acc + blin[cl];
}

// ---------------- launch ----------------
extern "C" void kernel_launch(void* const* d_in, const int* in_sizes, int n_in,
                              void* d_out, int out_size, void* d_ws, size_t ws_size,
                              hipStream_t stream) {
    const float* x     = (const float*)d_in[0];
    const int*   ei    = (const int*)d_in[1];
    const int*   src   = ei;
    const int*   dst   = ei + NE;
    const int*   batch = (const int*)d_in[2];
    const float* W1 = (const float*)d_in[3];  const float* b1 = (const float*)d_in[4];
    const float* W2 = (const float*)d_in[5];  const float* b2 = (const float*)d_in[6];
    const float* W3 = (const float*)d_in[7];  const float* b3 = (const float*)d_in[8];
    const float* Wl = (const float*)d_in[9];  const float* bl = (const float*)d_in[10];
    float* out = (float*)d_out;

    // workspace carve (256B aligned)
    char* ws = (char*)d_ws;
    size_t off = 0;
    auto carve = [&](size_t bytes) -> char* {
        char* p = ws + off;
        off = (off + bytes + 255) & ~(size_t)255;
        return p;
    };
    float*    deg  = (float*)carve(sizeof(float) * NN);
    float*    dinv = (float*)carve(sizeof(float) * NN);
    float*    norm = (float*)carve(sizeof(float) * NE);
    _Float16* xh   = (_Float16*)carve(sizeof(_Float16) * (size_t)NN * HID);
    float*    h    = (float*)carve(sizeof(float) * (size_t)NN * HID);
    float*    agg  = (float*)carve(sizeof(float) * (size_t)NN * HID);
    _Float16* bpak = (_Float16*)carve(sizeof(_Float16) * 8 * 16 * 32 * 16);
    float*    sums = (float*)carve(sizeof(float) * NGR * HID);
    float*    cnt  = (float*)carve(sizeof(float) * NGR);

    // 1) GCN normalization
    k_deg_init<<<(NN + 255) / 256, 256, 0, stream>>>(deg, NN);
    k_deg_acc <<<(NE + 255) / 256, 256, 0, stream>>>(dst, deg, NE);
    k_dinv    <<<(NN + 255) / 256, 256, 0, stream>>>(deg, dinv, NN);
    k_norm    <<<(NE + 255) / 256, 256, 0, stream>>>(src, dst, dinv, norm, NE);

    // 2) x -> f16
    {
        size_t n = (size_t)NN * IND;
        k_tohalf<<<(unsigned)((n + 255) / 256), 256, 0, stream>>>(x, xh, n, 0);
    }

    const int rowtiles = NN / 16;  // 6250 (exact)

    // ---- layer 1 (K=128) ----
    k_packB<<<((128 / 32) * 16 * 32 + 255) / 256, 256, 0, stream>>>(W1, bpak, 128);
    k_gemm_wmma<<<rowtiles, 128, 0, stream>>>(xh, bpak, h, 128);
    k_agg_init<<<NN, 256, 0, stream>>>(h, dinv, b1, agg);
    k_edge_agg<<<NE, 64, 0, stream>>>(src, dst, norm, h, agg);
    {
        size_t n = (size_t)NN * HID;
        k_tohalf<<<(unsigned)((n + 255) / 256), 256, 0, stream>>>(agg, xh, n, 1); // relu
    }

    // ---- layer 2 (K=256) ----
    k_packB<<<((256 / 32) * 16 * 32 + 255) / 256, 256, 0, stream>>>(W2, bpak, 256);
    k_gemm_wmma<<<rowtiles, 128, 0, stream>>>(xh, bpak, h, 256);
    k_agg_init<<<NN, 256, 0, stream>>>(h, dinv, b2, agg);
    k_edge_agg<<<NE, 64, 0, stream>>>(src, dst, norm, h, agg);
    {
        size_t n = (size_t)NN * HID;
        k_tohalf<<<(unsigned)((n + 255) / 256), 256, 0, stream>>>(agg, xh, n, 1); // relu
    }

    // ---- layer 3 (K=256, no relu) ----
    k_packB<<<((256 / 32) * 16 * 32 + 255) / 256, 256, 0, stream>>>(W3, bpak, 256);
    k_gemm_wmma<<<rowtiles, 128, 0, stream>>>(xh, bpak, h, 256);
    k_agg_init<<<NN, 256, 0, stream>>>(h, dinv, b3, agg);
    k_edge_agg<<<NE, 64, 0, stream>>>(src, dst, norm, h, agg);

    // 3) pool + linear
    k_pool_zero<<<(NGR * HID + 255) / 256, 256, 0, stream>>>(sums, cnt);
    k_pool_acc<<<(NN + 127) / 128, 256, 0, stream>>>(agg, batch, sums, cnt, NN);
    k_final<<<(NGR * NCLS + 255) / 256, 256, 0, stream>>>(sums, cnt, Wl, bl, out);
}
